// HSSBlock_72172630442132
// MI455X (gfx1250) — matrix operand order repaired and checked by reading, not compiled, not run on hardware
//
#include <hip/hip_runtime.h>
#include <stdint.h>

// ---------------- problem constants (from reference) ----------------
#define TOK    16384          // B*H*W tokens
#define CDIM   192
#define DIN    384            // EXPAND*C
#define E2     768            // 2*DIN
#define DSTATE 16
#define DTR    12             // (C+15)//16
#define KDIRS  4
#define LL     4096           // H*W
#define NCH    8              // scan chunks
#define CLEN   512            // LL / NCH

// ---------------- WMMA fragment types (gfx1250, wave32) ----------------
typedef __bf16        bf16x16 __attribute__((ext_vector_type(16)));
typedef float         f32x8   __attribute__((ext_vector_type(8)));
typedef unsigned int  u32v4   __attribute__((ext_vector_type(4)));

union FragU { bf16x16 bf; u32v4 q[2]; };

__device__ __forceinline__ uint16_t f2bf(float f) {       // RNE float->bf16 bits
  uint32_t u = __float_as_uint(f);
  uint32_t r = u + 0x7fffu + ((u >> 16) & 1u);
  return (uint16_t)(r >> 16);
}
__device__ __forceinline__ float silu_f(float x) { return x / (1.f + __expf(-x)); }
__device__ __forceinline__ float softplus_f(float x) {
  return (x > 20.f) ? x : log1pf(__expf(x));
}

// direction k, scan step l -> spatial token index s (= flips + HxW transpose)
__device__ __forceinline__ int tokmap(int k, int l) {
  int t = (k & 2) ? (LL - 1 - l) : l;
  return (k & 1) ? (((t & 63) << 6) | (t >> 6)) : t;
}

// A 16x32 bf16 fragment from row pointer: slot i -> k = (i%8) + 8*half + 16*(i/8)
__device__ __forceinline__ void loadA_frag(const uint16_t* row, int k0, int half, FragU& f) {
  const uint16_t* pa = row + k0 + 8 * half;
  f.q[0] = *(const u32v4*)pa;
  f.q[1] = *(const u32v4*)(pa + 16);
}
// B 32x16 bf16 fragment from column-row pointer: slot i -> k = (i%8) + 8*half + 16*(i/8)
__device__ __forceinline__ void loadB_frag(const uint16_t* row, int k0, int half, FragU& f) {
  const uint16_t* pb = row + k0 + 8 * half;
  f.q[0] = *(const u32v4*)pb;
  f.q[1] = *(const u32v4*)(pb + 16);
}

// ---------------- prep: bf16 weights, padded x_proj, A = -exp(A_logs) ----------------
__global__ void k_prep(const float* __restrict__ w_in, const float* __restrict__ w_out,
                       const float* __restrict__ w_xp, const float* __restrict__ A_logs,
                       uint16_t* __restrict__ w_in_bf, uint16_t* __restrict__ w_out_bf,
                       uint16_t* __restrict__ w48_bf, float* __restrict__ Aneg) {
  int i = blockIdx.x * blockDim.x + threadIdx.x;
  const int NIN = E2 * CDIM, NOUT = CDIM * DIN, NP = KDIRS * 48 * DIN, NA = KDIRS * DIN * DSTATE;
  if (i < NIN)  { w_in_bf[i]  = f2bf(w_in[i]);  return; }  i -= NIN;
  if (i < NOUT) { w_out_bf[i] = f2bf(w_out[i]); return; }  i -= NOUT;
  if (i < NP) {
    int k = i / (48 * DIN); int rem = i - k * 48 * DIN;
    int c = rem / DIN;      int d  = rem - c * DIN;
    float v = (c < 44) ? w_xp[(k * 44 + c) * DIN + d] : 0.f;
    w48_bf[i] = f2bf(v);  return;
  }  i -= NP;
  if (i < NA) { Aneg[i] = -__expf(A_logs[i]); }
}

// ---------------- LayerNorm over C=192, emit bf16 (one wave per token) ----------------
__global__ void k_ln1(const float* __restrict__ x, const float* __restrict__ g,
                      const float* __restrict__ b, uint16_t* __restrict__ xn_bf) {
  int lane = threadIdx.x & 31, wid = threadIdx.x >> 5;
  int tok  = blockIdx.x * 8 + wid;
  const float* row = x + (size_t)tok * CDIM;
  float v[6], s = 0.f, s2 = 0.f;
  #pragma unroll
  for (int j = 0; j < 6; j++) { v[j] = row[lane * 6 + j]; s += v[j]; s2 += v[j] * v[j]; }
  #pragma unroll
  for (int m = 16; m >= 1; m >>= 1) { s += __shfl_xor(s, m, 32); s2 += __shfl_xor(s2, m, 32); }
  float mu = s * (1.f / CDIM), var = s2 * (1.f / CDIM) - mu * mu;
  float rs = rsqrtf(var + 1e-6f);
  #pragma unroll
  for (int j = 0; j < 6; j++) {
    int c = lane * 6 + j;
    xn_bf[(size_t)tok * CDIM + c] = f2bf((v[j] - mu) * rs * g[c] + b[c]);
  }
}

// ---------------- bf16 WMMA GEMM, copy-free ping-pong over K (Kd % 64 == 0) ----------------
// C[M,N] = A[M,K] * Bw[N,K]^T (+resid); grid = (M/16, N/64), blockDim = 32
__global__ __launch_bounds__(32, 1)
void k_gemm_bf(const uint16_t* __restrict__ A, const uint16_t* __restrict__ Bw,
               const float* __restrict__ resid, float* __restrict__ Cfp,
               int N, int Kd) {
  int lane = threadIdx.x & 31;
  int m0 = blockIdx.x * 16, n0 = blockIdx.y * 64;
  int mlo = lane & 15, half = lane >> 4;
  const uint16_t* arow = A + (size_t)(m0 + mlo) * Kd;
  const uint16_t* brow[4];
  #pragma unroll
  for (int j = 0; j < 4; j++) brow[j] = Bw + (size_t)(n0 + 16 * j + mlo) * Kd;

  f32x8 acc[4] = {};
  FragU fa0, fb0[4], fa1, fb1[4];
  loadA_frag(arow, 0, half, fa0);
  #pragma unroll
  for (int j = 0; j < 4; j++) loadB_frag(brow[j], 0, half, fb0[j]);
  for (int k0 = 0; k0 < Kd; k0 += 64) {
    loadA_frag(arow, k0 + 32, half, fa1);              // buf1 loads in flight ...
    #pragma unroll
    for (int j = 0; j < 4; j++) loadB_frag(brow[j], k0 + 32, half, fb1[j]);
    #pragma unroll
    for (int j = 0; j < 4; j++)                        // ... while buf0 feeds the matrix pipe
      acc[j] = __builtin_amdgcn_wmma_f32_16x16x32_bf16(false, fa0.bf, false, fb0[j].bf,
                                                       (short)0, acc[j], false, false);
    int kn = k0 + 64;
    if (kn < Kd) {
      loadA_frag(arow, kn, half, fa0);
      #pragma unroll
      for (int j = 0; j < 4; j++) loadB_frag(brow[j], kn, half, fb0[j]);
    }
    #pragma unroll
    for (int j = 0; j < 4; j++)
      acc[j] = __builtin_amdgcn_wmma_f32_16x16x32_bf16(false, fa1.bf, false, fb1[j].bf,
                                                       (short)0, acc[j], false, false);
  }
  #pragma unroll
  for (int j = 0; j < 4; j++) {
    int n = n0 + 16 * j + mlo;
    #pragma unroll
    for (int r = 0; r < 8; r++) {                      // C/D: m = r + 8*half
      size_t off = (size_t)(m0 + r + 8 * half) * N + n;
      float rv = resid ? resid[off] : 0.f;
      Cfp[off] = acc[j][r] + rv;
    }
  }
}

// ---------------- depthwise 3x3 conv + bias + SiLU; dual-layout outputs ----------------
__global__ void k_conv(const float* __restrict__ xz, const float* __restrict__ cw,
                       const float* __restrict__ cb,
                       float* __restrict__ xc_f, uint16_t* __restrict__ xcT_bf) {
  int idx = blockIdx.x * blockDim.x + threadIdx.x;   // over 4*384*4096
  int b = idx / (DIN * LL); int rem = idx - b * DIN * LL;
  int d = rem / LL;         int s   = rem - d * LL;
  int h = s >> 6, w = s & 63;
  float acc = cb[d];
  #pragma unroll
  for (int kh = 0; kh < 3; kh++) {
    int hh = h + kh - 1; if ((unsigned)hh >= 64u) continue;
    #pragma unroll
    for (int kw = 0; kw < 3; kw++) {
      int ww = w + kw - 1; if ((unsigned)ww >= 64u) continue;
      int t = b * LL + hh * 64 + ww;
      acc += cw[d * 9 + kh * 3 + kw] * xz[(size_t)t * E2 + d];   // xv channels 0..383
    }
  }
  float v = silu_f(acc);
  xc_f [((size_t)b * DIN + d) * LL + s]  = v;          // [b,d,l] fp32 for the scan
  xcT_bf[((size_t)b * LL + s) * DIN + d] = f2bf(v);    // [b,l,d] bf16 for x_proj B-frags
}

__global__ void k_zero(float* p, int n) {
  int i = blockIdx.x * blockDim.x + threadIdx.x;
  if (i < n) p[i] = 0.f;
}

// ---------------- x_proj WMMA -> xdbl in [g][l][48] layout (scan-friendly) ----------------
// grid = (L/64, 48/16, b*k), blockDim = 32; B columns gathered via tokmap
__global__ __launch_bounds__(32, 1)
void k_xproj(const uint16_t* __restrict__ w48, const uint16_t* __restrict__ xcT,
             float* __restrict__ xdbl) {
  int lane = threadIdx.x & 31;
  int gk = blockIdx.z; int b = gk >> 2; int k = gk & 3;
  int m0 = blockIdx.y * 16, l0 = blockIdx.x * 64;
  int mlo = lane & 15, half = lane >> 4;
  const uint16_t* arow = w48 + (size_t)k * 48 * DIN + (size_t)(m0 + mlo) * DIN;
  const uint16_t* brow[4];
  #pragma unroll
  for (int j = 0; j < 4; j++)
    brow[j] = xcT + ((size_t)b * LL + tokmap(k, l0 + 16 * j + mlo)) * DIN;

  f32x8 acc[4] = {};
  FragU fa0, fb0[4], fa1, fb1[4];
  loadA_frag(arow, 0, half, fa0);
  #pragma unroll
  for (int j = 0; j < 4; j++) loadB_frag(brow[j], 0, half, fb0[j]);
  for (int k0 = 0; k0 < DIN; k0 += 64) {
    loadA_frag(arow, k0 + 32, half, fa1);
    #pragma unroll
    for (int j = 0; j < 4; j++) loadB_frag(brow[j], k0 + 32, half, fb1[j]);
    #pragma unroll
    for (int j = 0; j < 4; j++)
      acc[j] = __builtin_amdgcn_wmma_f32_16x16x32_bf16(false, fa0.bf, false, fb0[j].bf,
                                                       (short)0, acc[j], false, false);
    int kn = k0 + 64;
    if (kn < DIN) {
      loadA_frag(arow, kn, half, fa0);
      #pragma unroll
      for (int j = 0; j < 4; j++) loadB_frag(brow[j], kn, half, fb0[j]);
    }
    #pragma unroll
    for (int j = 0; j < 4; j++)
      acc[j] = __builtin_amdgcn_wmma_f32_16x16x32_bf16(false, fa1.bf, false, fb1[j].bf,
                                                       (short)0, acc[j], false, false);
  }
  float* out = xdbl + (size_t)gk * LL * 48;
  #pragma unroll
  for (int j = 0; j < 4; j++) {
    int l = l0 + 16 * j + mlo;
    #pragma unroll
    for (int r = 0; r < 8; r++)                          // [l][48]: 8 consecutive floats
      out[(size_t)l * 48 + m0 + r + 8 * half] = acc[j][r];
  }
}

// ================= chunked selective scan =================
// h_t = exp(dt*A)*h_{t-1} + dt*u*B_t ; chunk transition = exp(A * sum(dt)) (diagonal)
// Pass A: per-chunk local scan (h0=0) -> h_loc[16], sum(dt).  8x parallelism.
// Pass B: 8-step sequential prefix over chunks -> per-chunk start states.
// Pass C: per-chunk final scan emitting y, atomic merge of 4 directions.

// grid = 16g * 8ch * 3parts = 384 blocks, blockDim = 128
__global__ void k_scan_local(const float* __restrict__ xdbl, const float* __restrict__ xc_f,
                             const float* __restrict__ Aneg, const float* __restrict__ dtw_all,
                             const float* __restrict__ dtb_all,
                             float* __restrict__ hloc, float* __restrict__ dtsums) {
  int bid = blockIdx.x;
  int g = bid / (NCH * 3); int rem = bid - g * NCH * 3;
  int c = rem / 3;         int part = rem % 3;
  int d = part * 128 + threadIdx.x;
  int b = g >> 2, k = g & 3;
  const float* xd    = xdbl + (size_t)g * LL * 48;
  const float* xcrow = xc_f + ((size_t)b * DIN + d) * LL;
  float Ar[DSTATE], dtw[DTR], h[DSTATE];
  #pragma unroll
  for (int n = 0; n < DSTATE; n++) { Ar[n] = Aneg[((size_t)k * DIN + d) * DSTATE + n]; h[n] = 0.f; }
  #pragma unroll
  for (int r = 0; r < DTR; r++)      dtw[r] = dtw_all[((size_t)k * DIN + d) * DTR + r];
  float dtb = dtb_all[k * DIN + d];
  float dtsum = 0.f;
  for (int t = c * CLEN; t < (c + 1) * CLEN; t++) {
    __builtin_prefetch(xd + (size_t)(t + 4) * 48, 0, 1);
    float4 q[11];
    const float4* r4 = (const float4*)(xd + (size_t)t * 48);
    #pragma unroll
    for (int i = 0; i < 11; i++) q[i] = r4[i];
    const float* rv = (const float*)q;                   // rv[0..11]=dts, [12..27]=B, [28..43]=C
    float u  = xcrow[tokmap(k, t)];
    float dt = dtb;
    #pragma unroll
    for (int r = 0; r < DTR; r++) dt += dtw[r] * rv[r];
    dt = softplus_f(dt);
    dtsum += dt;
    float du = dt * u;
    #pragma unroll
    for (int n = 0; n < DSTATE; n++) h[n] = __expf(dt * Ar[n]) * h[n] + du * rv[DTR + n];
  }
  size_t base = (((size_t)g * DIN + d) * NCH + c) * DSTATE;
  #pragma unroll
  for (int n = 0; n < DSTATE; n++) hloc[base + n] = h[n];
  dtsums[((size_t)g * DIN + d) * NCH + c] = dtsum;
}

// grid = 48, blockDim = 128 : tiny sequential prefix over the 8 chunks
__global__ void k_scan_prefix(const float* __restrict__ hloc, const float* __restrict__ dtsums,
                              const float* __restrict__ Aneg, float* __restrict__ hstart) {
  int g = blockIdx.x / 3, part = blockIdx.x % 3;
  int d = part * 128 + threadIdx.x;
  int k = g & 3;
  float Ar[DSTATE], h[DSTATE];
  #pragma unroll
  for (int n = 0; n < DSTATE; n++) { Ar[n] = Aneg[((size_t)k * DIN + d) * DSTATE + n]; h[n] = 0.f; }
  for (int c = 0; c < NCH; c++) {
    size_t base = (((size_t)g * DIN + d) * NCH + c) * DSTATE;
    #pragma unroll
    for (int n = 0; n < DSTATE; n++) hstart[base + n] = h[n];
    float dtsum = dtsums[((size_t)g * DIN + d) * NCH + c];
    #pragma unroll
    for (int n = 0; n < DSTATE; n++) h[n] = __expf(Ar[n] * dtsum) * h[n] + hloc[base + n];
  }
}

// grid = 384, blockDim = 128 : final pass, emits y, atomic merge of directions
__global__ void k_scan_final(const float* __restrict__ xdbl, const float* __restrict__ xc_f,
                             const float* __restrict__ Aneg, const float* __restrict__ dtw_all,
                             const float* __restrict__ dtb_all, const float* __restrict__ Ds,
                             const float* __restrict__ hstart, float* __restrict__ ysum) {
  int bid = blockIdx.x;
  int g = bid / (NCH * 3); int rem = bid - g * NCH * 3;
  int c = rem / 3;         int part = rem % 3;
  int d = part * 128 + threadIdx.x;
  int b = g >> 2, k = g & 3;
  const float* xd    = xdbl + (size_t)g * LL * 48;
  const float* xcrow = xc_f + ((size_t)b * DIN + d) * LL;
  float Ar[DSTATE], dtw[DTR], h[DSTATE];
  size_t sbase = (((size_t)g * DIN + d) * NCH + c) * DSTATE;
  #pragma unroll
  for (int n = 0; n < DSTATE; n++) {
    Ar[n] = Aneg[((size_t)k * DIN + d) * DSTATE + n];
    h[n]  = hstart[sbase + n];
  }
  #pragma unroll
  for (int r = 0; r < DTR; r++) dtw[r] = dtw_all[((size_t)k * DIN + d) * DTR + r];
  float dtb = dtb_all[k * DIN + d];
  float Dv  = Ds[k * DIN + d];
  for (int t = c * CLEN; t < (c + 1) * CLEN; t++) {
    __builtin_prefetch(xd + (size_t)(t + 4) * 48, 0, 1);
    float4 q[11];
    const float4* r4 = (const float4*)(xd + (size_t)t * 48);
    #pragma unroll
    for (int i = 0; i < 11; i++) q[i] = r4[i];
    const float* rv = (const float*)q;
    int s = tokmap(k, t);
    float u  = xcrow[s];
    float dt = dtb;
    #pragma unroll
    for (int r = 0; r < DTR; r++) dt += dtw[r] * rv[r];
    dt = softplus_f(dt);
    float du = dt * u, y = 0.f;
    #pragma unroll
    for (int n = 0; n < DSTATE; n++) {
      h[n] = __expf(dt * Ar[n]) * h[n] + du * rv[DTR + n];
      y += h[n] * rv[DTR + DSTATE + n];
    }
    y += Dv * u;
    atomicAdd(&ysum[((size_t)b * LL + s) * DIN + d], y);   // merges all 4 directions
  }
}

// ---------------- out-LN over DIN + SiLU(z) gate, emit bf16 (one wave per token) ----------------
__global__ void k_merge(const float* __restrict__ ysum, const float* __restrict__ xz,
                        const float* __restrict__ g, const float* __restrict__ bb,
                        uint16_t* __restrict__ ygate_bf) {
  int lane = threadIdx.x & 31, wid = threadIdx.x >> 5;
  int tok  = blockIdx.x * 8 + wid;
  const float* row = ysum + (size_t)tok * DIN;
  float v[12], s = 0.f, s2 = 0.f;
  #pragma unroll
  for (int j = 0; j < 12; j++) { float t = row[j * 32 + lane]; v[j] = t; s += t; s2 += t * t; }
  #pragma unroll
  for (int m = 16; m >= 1; m >>= 1) { s += __shfl_xor(s, m, 32); s2 += __shfl_xor(s2, m, 32); }
  float mu = s * (1.f / DIN), var = s2 * (1.f / DIN) - mu * mu;
  float rs = rsqrtf(var + 1e-5f);
  #pragma unroll
  for (int j = 0; j < 12; j++) {
    int d = j * 32 + lane;
    float z  = xz[(size_t)tok * E2 + DIN + d];
    float yn = (v[j] - mu) * rs * g[d] + bb[d];
    ygate_bf[(size_t)tok * DIN + d] = f2bf(yn * silu_f(z));
  }
}

// ---------------- host launcher ----------------
static inline size_t algn256(size_t x) { return (x + 255) & ~(size_t)255; }

extern "C" void kernel_launch(void* const* d_in, const int* in_sizes, int n_in,
                              void* d_out, int out_size, void* d_ws, size_t ws_size,
                              hipStream_t stream) {
  const float* x_in   = (const float*)d_in[0];
  const float* ln1_g  = (const float*)d_in[1];
  const float* ln1_b  = (const float*)d_in[2];
  const float* w_in   = (const float*)d_in[3];
  const float* conv_w = (const float*)d_in[4];
  const float* conv_b = (const float*)d_in[5];
  const float* w_xp   = (const float*)d_in[6];
  const float* dt_w   = (const float*)d_in[7];
  const float* dt_b   = (const float*)d_in[8];
  const float* A_logs = (const float*)d_in[9];
  const float* Ds     = (const float*)d_in[10];
  const float* on_g   = (const float*)d_in[11];
  const float* on_b   = (const float*)d_in[12];
  const float* w_out  = (const float*)d_in[13];
  float* out = (float*)d_out;

  // workspace carve (~152 MB total)
  uint8_t* p = (uint8_t*)d_ws;
  auto take = [&](size_t bytes) { uint8_t* q = p; p += algn256(bytes); return q; };
  uint16_t* xn_bf    = (uint16_t*)take((size_t)TOK * CDIM * 2);
  uint16_t* w_in_bf  = (uint16_t*)take((size_t)E2 * CDIM * 2);
  uint16_t* w_out_bf = (uint16_t*)take((size_t)CDIM * DIN * 2);
  uint16_t* w48_bf   = (uint16_t*)take((size_t)KDIRS * 48 * DIN * 2);
  float*    Aneg     = (float*)   take((size_t)KDIRS * DIN * DSTATE * 4);
  float*    xz       = (float*)   take((size_t)TOK * E2 * 4);
  float*    xc_f     = (float*)   take((size_t)4 * DIN * LL * 4);
  uint16_t* xcT_bf   = (uint16_t*)take((size_t)TOK * DIN * 2);
  float*    xdbl     = (float*)   take((size_t)16 * LL * 48 * 4);
  float*    ysum     = (float*)   take((size_t)TOK * DIN * 4);
  uint16_t* ygate_bf = (uint16_t*)take((size_t)TOK * DIN * 2);
  float*    hloc     = (float*)   take((size_t)16 * DIN * NCH * DSTATE * 4);
  float*    hstart   = (float*)   take((size_t)16 * DIN * NCH * DSTATE * 4);
  float*    dtsums   = (float*)   take((size_t)16 * DIN * NCH * 4);

  // 1) weight prep (bf16 casts, x_proj zero-pad 44->48 rows, A = -exp(A_logs))
  {
    int total = E2 * CDIM + CDIM * DIN + KDIRS * 48 * DIN + KDIRS * DIN * DSTATE;
    k_prep<<<(total + 255) / 256, 256, 0, stream>>>(w_in, w_out, w_xp, A_logs,
                                                    w_in_bf, w_out_bf, w48_bf, Aneg);
  }
  // 2) LayerNorm1 -> bf16 activations
  k_ln1<<<TOK / 8, 256, 0, stream>>>(x_in, ln1_g, ln1_b, xn_bf);
  // 3) in_proj GEMM: [16384x192]*[192x768] -> xz fp32
  k_gemm_bf<<<dim3(TOK / 16, E2 / 64), 32, 0, stream>>>(xn_bf, w_in_bf, nullptr, xz, E2, CDIM);
  // 4) depthwise conv + SiLU
  k_conv<<<(4 * DIN * LL) / 256, 256, 0, stream>>>(xz, conv_w, conv_b, xc_f, xcT_bf);
  // 5) clear direction accumulator
  k_zero<<<(TOK * DIN + 255) / 256, 256, 0, stream>>>(ysum, TOK * DIN);
  // 6) x_proj GEMMs for all 4 scan directions (output in scan-friendly [l][48] layout)
  k_xproj<<<dim3(LL / 64, 3, 16), 32, 0, stream>>>(w48_bf, xcT_bf, xdbl);
  // 7) chunked selective scan: local -> prefix -> final (atomic merge of directions)
  k_scan_local <<<16 * NCH * 3, 128, 0, stream>>>(xdbl, xc_f, Aneg, dt_w, dt_b, hloc, dtsums);
  k_scan_prefix<<<48,           128, 0, stream>>>(hloc, dtsums, Aneg, hstart);
  k_scan_final <<<16 * NCH * 3, 128, 0, stream>>>(xdbl, xc_f, Aneg, dt_w, dt_b, Ds, hstart, ysum);
  // 8) out-LN + SiLU gate -> bf16
  k_merge<<<TOK / 8, 256, 0, stream>>>(ysum, xz, on_g, on_b, ygate_bf);
  // 9) out_proj GEMM + residual -> d_out
  k_gemm_bf<<<dim3(TOK / 16, CDIM / 64), 32, 0, stream>>>(ygate_bf, w_out_bf, x_in, out, CDIM, DIN);
}
